// KPFCNN_63170378990250
// MI455X (gfx1250) — matrix operand order repaired and checked by reading, not compiled, not run on hardware
//
#include <hip/hip_runtime.h>
#include <hip/hip_bf16.h>
#include <cstdint>
#include <cstddef>

// ============================================================================
// KPFCNN forward for gfx1250 (MI455X).
//  - All activations stored bf16 (halves HBM traffic; network is BW-bound).
//  - GEMMs: V_WMMA_F32_16X16X32_BF16, tiles staged via TENSOR_LOAD_TO_LDS
//    (TDM, double-buffered, s_wait_tensorcnt), LDS rows padded by TDM
//    (pad_interval=32 DW, pad_amount=1 DW -> 132B row stride, odd bank stride).
//  - fp32 accumulate everywhere; GroupNorm stats in fp32.
//
// Input flattening assumption (jax tree_flatten, dict keys sorted, None
// dropped):
//   0: feats  1..5: neighbors0..4  6..173: params  174..178: points0..4
//   179..182: subsampling0..3  183..186: upsampling0..3
//   params: 6..8 dec1{b,g,w} 9..11 dec2 12..14 dec3 15..17 dec4 18 desc_b
//   19 desc_w 20 det.b1 21 det.b2 22 det.w1 23 det.w2 24 det.w3
//   25..28 enc1_1{kp,kpw,nb,ng}; res blocks kp,kpw,nb,ng,[sc b,g,w],u1 b,g,w,
//   u2 b,g,w at: enc1_2@29(sc) enc2_1@42 enc2_2@52(sc) enc2_3@65 enc3_1@75
//   enc3_2@85(sc) enc3_3@98 enc4_1@108 enc4_2@118(sc) enc4_3@131 enc5_1@141
//   enc5_2@151(sc) enc5_3@164
// ============================================================================

typedef __attribute__((ext_vector_type(16))) __bf16 v16bf;
typedef __attribute__((ext_vector_type(8)))  float  v8f;
typedef __attribute__((ext_vector_type(4)))  unsigned int u32x4;
typedef __attribute__((ext_vector_type(8)))  int    i32x8;
typedef __attribute__((ext_vector_type(4)))  int    i32x4;

#define KPTS   15
#define NNBR   32
#define NGROUP 32

__device__ __forceinline__ float leakyf(float x) { return x >= 0.f ? x : 0.1f * x; }
__device__ __forceinline__ float softplusf(float x) {
  return fmaxf(x, 0.f) + log1pf(expf(-fabsf(x)));
}

// ---------------------------------------------------------------------------
// TDM 2D tile load: tile_dim0 = 64 elems (bf16), padded rows (+1 DWORD per
// 32 DWORDs) -> LDS row stride 66 bf16 = 132 B. OOB rows/cols read as zero.
// Descriptor per CDNA5 ISA ch.8 (D# groups).
// ---------------------------------------------------------------------------
#define GT_M 128
#define GT_N 64
#define GT_K 64
#define LDS_STRIDE 66   // bf16 elements per padded LDS row

__device__ __forceinline__ void tdm_load_2d(const void* gptr, unsigned lds_off,
                                            int rows_avail, int cols_avail,
                                            int stride_elems, int tile_rows) {
  unsigned long long ga = (unsigned long long)(uintptr_t)gptr;
  u32x4 g0;
  g0.x = 1u;                                             // count=1 (user D#)
  g0.y = lds_off;                                        // LDS byte address
  g0.z = (unsigned)ga;                                   // global addr lo32
  g0.w = (unsigned)((ga >> 32) & 0x01FFFFFFu) | (2u << 30); // addr[56:32]|type=2
  unsigned td0 = (unsigned)(cols_avail > 0 ? cols_avail : 0);
  unsigned td1 = (unsigned)(rows_avail > 0 ? rows_avail : 0);
  i32x8 g1;
  // data_size=1 (2B), pad_enable, pad_interval=4 (32 DWORDs), pad_amount=0 (1 DW)
  g1[0] = (int)((1u << 16) | (1u << 20) | (4u << 22));
  g1[1] = (int)((td0 & 0xFFFFu) << 16);                  // tensor_dim0[15:0]
  g1[2] = (int)((td0 >> 16) | ((td1 & 0xFFFFu) << 16));  // td0 hi | td1 lo
  g1[3] = (int)((td1 >> 16) | ((unsigned)GT_K << 16));   // td1 hi | tile_dim0
  g1[4] = (int)(unsigned)tile_rows;                      // tile_dim1 (dim2=0)
  g1[5] = (int)(unsigned)stride_elems;                   // dim0 stride lo32
  g1[6] = 0;                                             // stride hi, d1stride lo
  g1[7] = 0;
  i32x4 gz = {};
#if defined(__clang_major__) && (__clang_major__ >= 23)
  i32x8 gz8 = {};
  __builtin_amdgcn_tensor_load_to_lds(g0, g1, gz, gz, gz8, 0);
#else
  __builtin_amdgcn_tensor_load_to_lds(g0, g1, gz, gz, 0);
#endif
}

// ---------------------------------------------------------------------------
// bf16 WMMA GEMM: C[M,N] = A[M,K] * BT[N,K]^T   (BT is pre-transposed bf16)
// Block tile 128x64, 8 waves (wave = 32x32 region = 2x2 WMMA tiles),
// K staged in 64-deep chunks by TDM, double buffered.
// act: 0=none 1=leaky 2=softplus. cnt: per-row divisor. c_f32: output dtype.
// ---------------------------------------------------------------------------
__global__ __launch_bounds__(256)
void gemm_bf16_wmma(const __bf16* __restrict__ A, const __bf16* __restrict__ BT,
                    void* __restrict__ Cv, int M, int N, int K,
                    const float* __restrict__ cnt, const float* __restrict__ bias,
                    int act, int c_f32) {
  __shared__ __bf16 Asb[2][GT_M * LDS_STRIDE];
  __shared__ __bf16 Bsb[2][GT_N * LDS_STRIDE];
  const int tid  = threadIdx.x;
  const int lane = tid & 31;
  const int wave = tid >> 5;
  const int wm   = wave >> 1;       // 0..3
  const int wn   = wave & 1;        // 0..1
  const int bm   = blockIdx.x * GT_M;
  const int bn   = blockIdx.y * GT_N;
  const int mrow = lane & 15;
  const int hi   = lane >> 4;       // selects K+8 / K+24 halves (ISA 7.12.2)
  const int nchunk = (K + GT_K - 1) / GT_K;

  if (wave == 0) {
    tdm_load_2d(A + (size_t)bm * K, (unsigned)(uintptr_t)&Asb[0][0],
                M - bm, K, K, GT_M);
    tdm_load_2d(BT + (size_t)bn * K, (unsigned)(uintptr_t)&Bsb[0][0],
                N - bn, K, K, GT_N);
  }

  v8f zero = {};
  v8f acc[2][2];
  acc[0][0] = zero; acc[0][1] = zero; acc[1][0] = zero; acc[1][1] = zero;

  for (int t = 0; t < nchunk; ++t) {
    if (wave == 0) {
      if (t + 1 < nchunk) {
        int k0 = (t + 1) * GT_K;
        tdm_load_2d(A + (size_t)bm * K + k0, (unsigned)(uintptr_t)&Asb[(t + 1) & 1][0],
                    M - bm, K - k0, K, GT_M);
        tdm_load_2d(BT + (size_t)bn * K + k0, (unsigned)(uintptr_t)&Bsb[(t + 1) & 1][0],
                    N - bn, K - k0, K, GT_N);
        __builtin_amdgcn_s_wait_tensorcnt(2);  // oldest 2 (chunk t) complete
      } else {
        __builtin_amdgcn_s_wait_tensorcnt(0);
      }
    }
    __syncthreads();

    const __bf16* Ar = &Asb[t & 1][0];
    const __bf16* Br = &Bsb[t & 1][0];
    #pragma unroll
    for (int kk = 0; kk < GT_K; kk += 32) {
      v16bf af[2], bfm[2];
      #pragma unroll
      for (int i = 0; i < 2; ++i) {
        const __bf16* ra = Ar + (size_t)(wm * 32 + i * 16 + mrow) * LDS_STRIDE + kk;
        #pragma unroll
        for (int e = 0; e < 8; ++e) {
          af[i][e]     = ra[hi * 8 + e];        // K = kk + hi*8 + e
          af[i][8 + e] = ra[16 + hi * 8 + e];   // K = kk + 16 + hi*8 + e
        }
        const __bf16* rb = Br + (size_t)(wn * 32 + i * 16 + mrow) * LDS_STRIDE + kk;
        #pragma unroll
        for (int e = 0; e < 8; ++e) {
          bfm[i][e]     = rb[hi * 8 + e];        // K = kk + hi*8 + e
          bfm[i][8 + e] = rb[16 + hi * 8 + e];   // K = kk + 16 + hi*8 + e
        }
      }
      #pragma unroll
      for (int i = 0; i < 2; ++i)
        #pragma unroll
        for (int j = 0; j < 2; ++j)
          acc[i][j] = __builtin_amdgcn_wmma_f32_16x16x32_bf16(
              false, af[i], false, bfm[j], (short)0, acc[i][j], false, false);
    }
    __syncthreads();
  }

  // Epilogue (C/D layout: VGPR r -> M = r + 8*hi, N = lane&15)
  #pragma unroll
  for (int i = 0; i < 2; ++i) {
    #pragma unroll
    for (int j = 0; j < 2; ++j) {
      #pragma unroll
      for (int r = 0; r < 8; ++r) {
        int gm = bm + wm * 32 + i * 16 + r + 8 * hi;
        int gn = bn + wn * 32 + j * 16 + (lane & 15);
        if (gm < M && gn < N) {
          float v = acc[i][j][r];
          if (cnt)  v /= cnt[gm];
          if (bias) v += bias[gn];
          if (act == 1)      v = leakyf(v);
          else if (act == 2) v = softplusf(v);
          if (c_f32) ((float*)Cv)[(size_t)gm * N + gn] = v;
          else       ((__bf16*)Cv)[(size_t)gm * N + gn] = (__bf16)v;
        }
      }
    }
  }
}

// ---------------------------------------------------------------------------
// Weight convert: fp32 W[K,N] -> bf16 W^T[N,K] (contiguous rows for TDM).
// ---------------------------------------------------------------------------
__global__ __launch_bounds__(256)
void wt_convert(const float* __restrict__ W, __bf16* __restrict__ WT,
                int K, int N) {
  long long total = (long long)K * N;
  for (long long i = (long long)blockIdx.x * 256 + threadIdx.x; i < total;
       i += (long long)gridDim.x * 256) {
    int n = (int)(i / K), k = (int)(i % K);
    WT[i] = (__bf16)W[(size_t)k * N + n];
  }
}

__global__ __launch_bounds__(256)
void f32_to_bf16(const float* __restrict__ in, __bf16* __restrict__ out,
                 long long total) {
  for (long long i = (long long)blockIdx.x * 256 + threadIdx.x; i < total;
       i += (long long)gridDim.x * 256)
    out[i] = (__bf16)in[i];
}

// ---------------------------------------------------------------------------
// KPConv phase 1: one wave32 per query point (lane == neighbor, NN==32).
// weighted[Nq, KPTS*Ci] (bf16) and valid-neighbor count (fp32).
// ---------------------------------------------------------------------------
__global__ __launch_bounds__(32)
void kpconv_weighted_kernel(const __bf16* __restrict__ sfeats,
                            const float* __restrict__ qpts,
                            const float* __restrict__ spts,
                            const int* __restrict__ nidx,
                            const float* __restrict__ kpts,
                            float inv_sigma,
                            __bf16* __restrict__ weighted,
                            float* __restrict__ cntv,
                            int Nq, int Ns, int Ci) {
  __shared__ float wlds[KPTS][NNBR];
  __shared__ float xlds[NNBR][NNBR + 1];
  const int n = blockIdx.x;
  const int lane = threadIdx.x;

  const int idx = nidx[(size_t)n * NNBR + lane];
  const bool valid = (idx >= 0 && idx < Ns);

  const float qx = qpts[n * 3 + 0], qy = qpts[n * 3 + 1], qz = qpts[n * 3 + 2];
  float px, py, pz;
  if (valid) {
    px = spts[(size_t)idx * 3 + 0];
    py = spts[(size_t)idx * 3 + 1];
    pz = spts[(size_t)idx * 3 + 2];
    __builtin_prefetch(sfeats + (size_t)idx * Ci, 0, 0);  // global_prefetch_b8
  } else {
    px = py = pz = 1.0e6f;  // sentinel (far away -> weight clamps to 0)
  }
  const float dxq = px - qx, dyq = py - qy, dzq = pz - qz;

  #pragma unroll
  for (int k = 0; k < KPTS; ++k) {
    float dx = dxq - kpts[k * 3 + 0];
    float dy = dyq - kpts[k * 3 + 1];
    float dz = dzq - kpts[k * 3 + 2];
    float d = sqrtf(dx * dx + dy * dy + dz * dz);
    wlds[k][lane] = fmaxf(1.0f - d * inv_sigma, 0.0f);
  }

  float rsum = 0.f;
  const size_t wbase = (size_t)n * (KPTS * (size_t)Ci);

  for (int c0 = 0; c0 < Ci; c0 += NNBR) {
    const int cw = (Ci - c0 < NNBR) ? (Ci - c0) : NNBR;
    for (int a = 0; a < NNBR; ++a) {
      int ia = __shfl(idx, a, 32);
      float v = 0.f;
      if (ia >= 0 && ia < Ns && lane < cw) v = (float)sfeats[(size_t)ia * Ci + c0 + lane];
      xlds[a][lane] = v;
    }
    __syncthreads();  // single-wave workgroup -> S_NOP; orders LDS

    for (int c = 0; c < cw; ++c) rsum += xlds[lane][c];

    #pragma unroll 3
    for (int k = 0; k < KPTS; ++k) {
      float accv = 0.f;
      #pragma unroll 8
      for (int a = 0; a < NNBR; ++a) accv += wlds[k][a] * xlds[a][lane];
      if (lane < cw) weighted[wbase + (size_t)k * Ci + c0 + lane] = (__bf16)accv;
    }
    __syncthreads();
  }

  unsigned long long bm = __ballot(rsum > 0.0f);
  int cc = __popcll(bm);
  if (lane == 0) cntv[n] = (float)(cc > 0 ? cc : 1);
}

// ---------------------------------------------------------------------------
// GroupNorm (fp32 stats over bf16 activations): partials -> finalize -> apply
// ---------------------------------------------------------------------------
#define GN_RB 64

__global__ __launch_bounds__(256)
void gn_partial(const __bf16* __restrict__ x, float* __restrict__ part,
                int N, int C, int rpb) {
  const int cg = C / NGROUP;
  const int c = blockIdx.y * 256 + threadIdx.x;
  const int r0 = blockIdx.x * rpb;
  int r1 = r0 + rpb; if (r1 > N) r1 = N;
  float s = 0.f, sq = 0.f;
  if (c < C) {
    for (int n = r0; n < r1; ++n) {
      float v = (float)x[(size_t)n * C + c];
      s += v; sq += v * v;
    }
  }
  __shared__ float ss[256], sv[256];
  ss[threadIdx.x] = s; sv[threadIdx.x] = sq;
  __syncthreads();
  if (c < C && (threadIdx.x % cg) == 0) {
    float a = 0.f, b = 0.f;
    for (int j = 0; j < cg; ++j) { a += ss[threadIdx.x + j]; b += sv[threadIdx.x + j]; }
    int g = c / cg;
    part[(g * GN_RB + blockIdx.x) * 2 + 0] = a;
    part[(g * GN_RB + blockIdx.x) * 2 + 1] = b;
  }
}

__global__ __launch_bounds__(32)
void gn_finalize(const float* __restrict__ part, float* __restrict__ stats,
                 int N, int C) {
  int g = threadIdx.x;
  if (g >= NGROUP) return;
  float s = 0.f, sq = 0.f;
  for (int r = 0; r < GN_RB; ++r) {
    s += part[(g * GN_RB + r) * 2 + 0];
    sq += part[(g * GN_RB + r) * 2 + 1];
  }
  float cnt = (float)N * (float)(C / NGROUP);
  float mean = s / cnt;
  float var = sq / cnt - mean * mean;
  stats[g * 2 + 0] = mean;
  stats[g * 2 + 1] = rsqrtf(var + 1e-5f);
}

__global__ __launch_bounds__(256)
void gn_apply(__bf16* __restrict__ x, const float* __restrict__ stats,
              const float* __restrict__ gamma, const float* __restrict__ beta,
              long long total, int C, int act) {
  const int cg = C / NGROUP;
  for (long long i = (long long)blockIdx.x * 256 + threadIdx.x; i < total;
       i += (long long)gridDim.x * 256) {
    int c = (int)(i % C);
    int g = c / cg;
    float v = ((float)x[i] - stats[2 * g]) * stats[2 * g + 1] * gamma[c] + beta[c];
    if (act == 1) v = leakyf(v);
    x[i] = (__bf16)v;
  }
}

// ---------------------------------------------------------------------------
// Misc elementwise kernels (bf16 activations, fp32 math)
// ---------------------------------------------------------------------------
__global__ __launch_bounds__(256)
void max_gather(const __bf16* __restrict__ in, const int* __restrict__ nidx,
                __bf16* __restrict__ out, int Nq, int Ns, int C) {
  long long total = (long long)Nq * C;
  for (long long i = (long long)blockIdx.x * 256 + threadIdx.x; i < total;
       i += (long long)gridDim.x * 256) {
    int n = (int)(i / C), c = (int)(i % C);
    const int* nb = nidx + (size_t)n * NNBR;
    float m = -3.4e38f;
    #pragma unroll 4
    for (int a = 0; a < NNBR; ++a) {
      int id = nb[a];
      float v = (id >= 0 && id < Ns) ? (float)in[(size_t)id * C + c] : 0.f;
      m = fmaxf(m, v);
    }
    out[i] = (__bf16)m;
  }
}

__global__ __launch_bounds__(256)
void upsample_concat(const __bf16* __restrict__ fnext, const __bf16* __restrict__ fcur,
                     const int* __restrict__ ups, __bf16* __restrict__ out,
                     int N, int Cup, int Ccur, int Nn) {
  const int Ct = Cup + Ccur;
  long long total = (long long)N * Ct;
  for (long long i = (long long)blockIdx.x * 256 + threadIdx.x; i < total;
       i += (long long)gridDim.x * 256) {
    int n = (int)(i / Ct), c = (int)(i % Ct);
    __bf16 v;
    if (c < Cup) {
      int id = ups[n];
      v = (id >= 0 && id < Nn) ? fnext[(size_t)id * Cup + c] : (__bf16)0.f;
    } else {
      v = fcur[(size_t)n * Ccur + (c - Cup)];
    }
    out[i] = v;
  }
}

__global__ __launch_bounds__(256)
void add_leaky(const __bf16* __restrict__ a, const __bf16* __restrict__ b,
               __bf16* __restrict__ out, long long total) {
  for (long long i = (long long)blockIdx.x * 256 + threadIdx.x; i < total;
       i += (long long)gridDim.x * 256)
    out[i] = (__bf16)leakyf((float)a[i] + (float)b[i]);
}

// ===========================================================================
// Host orchestration
// ===========================================================================
namespace {
struct RBIdx { int kp, kpw, nb, ng, scb, scg, scw, u1b, u1g, u1w, u2b, u2g, u2w; };
inline RBIdx mk_sc(int b)   { return {b, b+1, b+2, b+3, b+4, b+5, b+6, b+7, b+8, b+9, b+10, b+11, b+12}; }
inline RBIdx mk_nosc(int b) { return {b, b+1, b+2, b+3, -1, -1, -1, b+4, b+5, b+6, b+7, b+8, b+9}; }
inline int gsz(long long t) { long long b = (t + 255) / 256; return (int)(b > 262144 ? 262144 : b); }
}

extern "C" void kernel_launch(void* const* d_in, const int* in_sizes, int n_in,
                              void* d_out, int out_size, void* d_ws, size_t ws_size,
                              hipStream_t stream) {
  (void)in_sizes; (void)n_in; (void)out_size; (void)ws_size;
  auto PF = [&](int i) { return (const float*)d_in[i]; };
  auto PI = [&](int i) { return (const int*)d_in[i]; };

  static const int NS[5] = {40000, 10000, 2500, 640, 160};
  const float SIG[5] = {0.6f, 1.2f, 2.4f, 4.8f, 9.6f};
  const float* PTS[5]; const int* NBR[5]; const int* SUB[4]; const int* UPS[4];
  for (int i = 0; i < 5; ++i) { PTS[i] = PF(174 + i); NBR[i] = PI(1 + i); }
  for (int i = 0; i < 4; ++i) { SUB[i] = PI(179 + i); UPS[i] = PI(183 + i); }

  // ---- workspace (byte bump allocator, 256B aligned) ----
  char* base = (char*)d_ws;
  size_t off = 0;
  auto allocB = [&](size_t bytes) { void* p = base + off; off += (bytes + 255) & ~(size_t)255; return p; };
  auto allocH = [&](size_t elems) { return (__bf16*)allocB(elems * 2); };
  auto allocF = [&](size_t elems) { return (float*)allocB(elems * 4); };

  __bf16* f1   = allocH(40000LL * 128);
  __bf16* f2   = allocH(10000LL * 256);
  __bf16* f3   = allocH(2500LL * 512);
  __bf16* f4   = allocH(640LL * 1024);
  __bf16* f5   = allocH(160LL * 2048);
  __bf16* bufA = allocH(40000LL * 64);
  __bf16* Sw   = allocH(40000LL * (KPTS * 32));   // weighted / concat scratch
  __bf16* St1  = allocH(40000LL * 32);
  __bf16* St2  = allocH(40000LL * 32);
  __bf16* St3  = allocH(40000LL * 128);
  __bf16* Sc0  = allocH(10000LL * 128);
  __bf16* Sc1  = allocH(40000LL * 128);
  __bf16* Fbf  = allocH(40000LL);                 // feats in bf16
  __bf16* Wbf  = allocH((size_t)KPTS * 512 * 512); // transposed weight scratch
  float*  Scnt  = allocF(40000);
  float*  Gpart = allocF(NGROUP * GN_RB * 2);
  float*  Gstat = allocF(NGROUP * 2);

  // ---- helpers ----
  auto gemm = [&](const __bf16* A, const float* W, void* C, int M, int N, int K,
                  const float* cnt, const float* bias, int act, int c_f32) {
    wt_convert<<<gsz((long long)K * N), 256, 0, stream>>>(W, Wbf, K, N);
    dim3 g((M + GT_M - 1) / GT_M, (N + GT_N - 1) / GT_N);
    gemm_bf16_wmma<<<g, 256, 0, stream>>>(A, Wbf, C, M, N, K, cnt, bias, act, c_f32);
  };
  auto gnorm = [&](__bf16* X, int N, int C, const float* gamma, const float* beta, int act) {
    int rpb = (N + GN_RB - 1) / GN_RB;
    dim3 g1(GN_RB, (C + 255) / 256);
    gn_partial<<<g1, 256, 0, stream>>>(X, Gpart, N, C, rpb);
    gn_finalize<<<1, 32, 0, stream>>>(Gpart, Gstat, N, C);
    long long tot = (long long)N * C;
    gn_apply<<<gsz(tot), 256, 0, stream>>>(X, Gstat, gamma, beta, tot, C, act);
  };
  auto kpconv = [&](const __bf16* feats, const float* qp, const float* sp,
                    const int* nid, const float* kp, float sigma,
                    int Nq, int Ns_, int Ci) {
    kpconv_weighted_kernel<<<Nq, 32, 0, stream>>>(feats, qp, sp, nid, kp,
                                                  1.0f / sigma, Sw, Scnt, Nq, Ns_, Ci);
  };

  auto res_block = [&](const RBIdx& ix, const __bf16* fin, __bf16* fout,
                       int Nq, int Ns_, int ci, int co,
                       const float* qp, const float* sp, const int* nid,
                       float sigma, bool strided) {
    const int c4 = co / 4;
    gemm(fin, PF(ix.u1w), St1, Ns_, c4, ci, nullptr, nullptr, 0, 0);
    gnorm(St1, Ns_, c4, PF(ix.u1g), PF(ix.u1b), 1);
    kpconv(St1, qp, sp, nid, PF(ix.kp), sigma, Nq, Ns_, c4);
    gemm(Sw, PF(ix.kpw), St2, Nq, c4, KPTS * c4, Scnt, nullptr, 0, 0);
    gnorm(St2, Nq, c4, PF(ix.ng), PF(ix.nb), 1);
    gemm(St2, PF(ix.u2w), St3, Nq, co, c4, nullptr, nullptr, 0, 0);
    gnorm(St3, Nq, co, PF(ix.u2g), PF(ix.u2b), 0);
    const __bf16* scin = fin;
    if (strided) {
      max_gather<<<gsz((long long)Nq * ci), 256, 0, stream>>>(fin, nid, Sc0, Nq, Ns_, ci);
      scin = Sc0;
    }
    const __bf16* scf = scin;
    if (ix.scw >= 0) {
      gemm(scin, PF(ix.scw), Sc1, Nq, co, ci, nullptr, nullptr, 0, 0);
      gnorm(Sc1, Nq, co, PF(ix.scg), PF(ix.scb), 0);
      scf = Sc1;
    }
    add_leaky<<<gsz((long long)Nq * co), 256, 0, stream>>>(St3, scf, fout, (long long)Nq * co);
  };

  // ---- encoder ----
  f32_to_bf16<<<gsz(40000), 256, 0, stream>>>(PF(0), Fbf, 40000);
  kpconv(Fbf, PTS[0], PTS[0], NBR[0], PF(25), SIG[0], NS[0], NS[0], 1);
  gemm(Sw, PF(26), bufA, NS[0], 64, KPTS * 1, Scnt, nullptr, 0, 0);
  gnorm(bufA, NS[0], 64, PF(28), PF(27), 1);  // gamma=ng, beta=nb

  res_block(mk_sc(29),    bufA, f1, NS[0], NS[0], 64,   128,  PTS[0], PTS[0], NBR[0], SIG[0], false);
  res_block(mk_nosc(42),  f1,   f2, NS[1], NS[0], 128,  128,  PTS[1], PTS[0], SUB[0], SIG[0], true);
  res_block(mk_sc(52),    f2,   f2, NS[1], NS[1], 128,  256,  PTS[1], PTS[1], NBR[1], SIG[1], false);
  res_block(mk_nosc(65),  f2,   f2, NS[1], NS[1], 256,  256,  PTS[1], PTS[1], NBR[1], SIG[1], false);
  res_block(mk_nosc(75),  f2,   f3, NS[2], NS[1], 256,  256,  PTS[2], PTS[1], SUB[1], SIG[1], true);
  res_block(mk_sc(85),    f3,   f3, NS[2], NS[2], 256,  512,  PTS[2], PTS[2], NBR[2], SIG[2], false);
  res_block(mk_nosc(98),  f3,   f3, NS[2], NS[2], 512,  512,  PTS[2], PTS[2], NBR[2], SIG[2], false);
  res_block(mk_nosc(108), f3,   f4, NS[3], NS[2], 512,  512,  PTS[3], PTS[2], SUB[2], SIG[2], true);
  res_block(mk_sc(118),   f4,   f4, NS[3], NS[3], 512,  1024, PTS[3], PTS[3], NBR[3], SIG[3], false);
  res_block(mk_nosc(131), f4,   f4, NS[3], NS[3], 1024, 1024, PTS[3], PTS[3], NBR[3], SIG[3], false);
  res_block(mk_nosc(141), f4,   f5, NS[4], NS[3], 1024, 1024, PTS[4], PTS[3], SUB[3], SIG[3], true);
  res_block(mk_sc(151),   f5,   f5, NS[4], NS[4], 1024, 2048, PTS[4], PTS[4], NBR[4], SIG[4], false);
  res_block(mk_nosc(164), f5,   f5, NS[4], NS[4], 2048, 2048, PTS[4], PTS[4], NBR[4], SIG[4], false);

  // ---- decoder (cat in Sw, outputs ping-pong St3/Sc1) ----
  auto dec = [&](const __bf16* fnext, int Nn, int Cup, const __bf16* fcur, int N, int Ccur,
                 const int* ups, int wB, int wG, int wW, __bf16* gout, int Cout) {
    int Ct = Cup + Ccur;
    upsample_concat<<<gsz((long long)N * Ct), 256, 0, stream>>>(fnext, fcur, ups, Sw, N, Cup, Ccur, Nn);
    gemm(Sw, PF(wW), gout, N, Cout, Ct, nullptr, nullptr, 0, 0);
    gnorm(gout, N, Cout, PF(wG), PF(wB), 1);
  };
  dec(f5,  NS[4], 2048, f4, NS[3], 1024, UPS[3], 15, 16, 17, St3, 1024);
  dec(St3, NS[3], 1024, f3, NS[2], 512,  UPS[2], 12, 13, 14, Sc1, 512);
  dec(Sc1, NS[2], 512,  f2, NS[1], 256,  UPS[1], 9, 10, 11,  St3, 256);
  dec(St3, NS[1], 256,  f1, NS[0], 128,  UPS[0], 6, 7, 8,    Sc1, 128);

  // ---- heads ----
  float* out = (float*)d_out;
  gemm(Sc1, PF(19), out, NS[0], 32, 128, nullptr, PF(18), 0, 1);                 // desc
  gemm(Sc1, PF(22), St3, NS[0], 64, 128, nullptr, PF(20), 2, 0);                 // det h1
  gemm(St3, PF(23), St2, NS[0], 32, 64,  nullptr, PF(21), 2, 0);                 // det h2
  gemm(St2, PF(24), out + (size_t)NS[0] * 32, NS[0], 1, 32, nullptr, nullptr, 2, 1); // det
}
